// SpellerModel_18021682774505
// MI455X (gfx1250) — matrix operand order repaired and checked by reading, not compiled, not run on hardware
//
#include <hip/hip_runtime.h>
#include <hip/hip_bf16.h>
#include <cstddef>

// ---------------------------------------------------------------------------
// Speller (attention LSTM decoder) for MI455X / gfx1250, wave32 + WMMA bf16.
//
// H=256, CTX=128, Q=128, V=33, L=3, B=64, T=2048, S=220.
// Dense per-step GEMMs run on v_wmma_f32_16x16x32_bf16 (one wave per 16x16
// C tile). Attention energy/context are bandwidth-bound matvecs (key+value =
// 128MB, resident in the 192MB L2) and use plain f32 VALU dot products.
// ---------------------------------------------------------------------------

#define HB 256
#define CTXB 128
#define QB 128
#define VB 33
#define BB 64
#define TB 2048
#define SB 220

typedef __attribute__((ext_vector_type(16))) __bf16 v16bf;
typedef __attribute__((ext_vector_type(8)))  float  v8f;

union Frag16 { v16bf v; unsigned int w[8]; };

__device__ __forceinline__ unsigned short f2bf(float f) {
  unsigned int u = __float_as_uint(f);
  u = (u + 0x7FFFu + ((u >> 16) & 1u)) >> 16;   // round-to-nearest-even
  return (unsigned short)u;
}

// A fragment: 16x32 (MxK) bf16, row-major source, ISA 7.12.2 layout.
// lanes 0-15: K {0..7,16..23}; lanes 16-31: K {8..15,24..31}; row = lane&15.
__device__ __forceinline__ v16bf load_frag_a(const unsigned short* A, int lda, int lane) {
  Frag16 f;
  const unsigned short* base = A + (lane & 15) * lda + ((lane >> 4) << 3);
#pragma unroll
  for (int i = 0; i < 4; ++i) {
    f.w[i]     = *(const unsigned int*)(base + 2 * i);
    f.w[i + 4] = *(const unsigned int*)(base + 16 + 2 * i);
  }
  return f.v;
}

// B fragment: 32x16 (KxN) bf16 where B[k][n] = W[n][k], W row-major [N][K].
// lanes 0-15: K {0..7,16..23}; lanes 16-31: K {8..15,24..31}; col = lane&15. K-pairs contiguous.
__device__ __forceinline__ v16bf load_frag_b(const unsigned short* W, int ldk, int lane) {
  Frag16 f;
  const unsigned short* base = W + (lane & 15) * ldk + ((lane >> 4) << 3);
#pragma unroll
  for (int i = 0; i < 8; ++i) f.w[i] = *(const unsigned int*)(base + 2 * (i & 3) + ((i >> 2) << 4));
  return f.v;
}

// C = A(64xK, bf16) * W^T (W is [N][K] bf16) + bias, optional ReLU.
// One wave per 16x16 output tile. blockDim=128 (4 waves), grid = ntiles blocks
// (M is always 64 -> 4 M-tiles, so waves = 4*ntiles).
__global__ void gemm_bf16_wmma(const unsigned short* __restrict__ A, int lda,
                               const unsigned short* __restrict__ W, int ldk,
                               float* __restrict__ C, int ldc,
                               const float* __restrict__ bias,
                               int ntiles, int K, int relu) {
  int wid  = (blockIdx.x * blockDim.x + threadIdx.x) >> 5;
  int lane = threadIdx.x & 31;
  int mt = wid / ntiles;
  int nt = wid % ntiles;
  const unsigned short* Ab = A + mt * 16 * lda;
  const unsigned short* Wb = W + nt * 16 * ldk;
  v8f acc = {};
  for (int k = 0; k < K; k += 32) {
    v16bf a = load_frag_a(Ab + k, lda, lane);
    v16bf b = load_frag_b(Wb + k, ldk, lane);
    acc = __builtin_amdgcn_wmma_f32_16x16x32_bf16(false, a, false, b,
                                                  (short)0, acc, false, false);
  }
  int col  = nt * 16 + (lane & 15);
  int row0 = mt * 16 + ((lane >> 4) << 3);
  float bv = bias ? bias[col] : 0.f;
#pragma unroll
  for (int i = 0; i < 8; ++i) {
    float v = acc[i] + bv;
    if (relu) v = fmaxf(v, 0.f);
    C[(size_t)(row0 + i) * ldc + col] = v;
  }
}

// LSTM cell pointwise update; gates = [i|f|g|o] (bias already added in GEMM).
// Writes h (bf16) into two destination slots (self-state for next step, and
// next layer's / feat buffer input for this step).
__global__ void lstm_cell_kernel(const float* __restrict__ gates,  // [64][1024]
                                 float* __restrict__ cstate,       // [64][256]
                                 unsigned short* __restrict__ hdst0, int ld0, int off0,
                                 unsigned short* __restrict__ hdst1, int ld1, int off1) {
  int idx = blockIdx.x * blockDim.x + threadIdx.x;  // 64*256
  int b = idx >> 8, u = idx & 255;
  const float* g = gates + b * 1024;
  float gi = g[u], gf = g[u + 256], gg = g[u + 512], go = g[u + 768];
  float si = 1.f / (1.f + __expf(-gi));
  float sf = 1.f / (1.f + __expf(-gf));
  float so = 1.f / (1.f + __expf(-go));
  float cn = sf * cstate[idx] + si * tanhf(gg);
  cstate[idx] = cn;
  unsigned short hb = f2bf(so * tanhf(cn));
  hdst0[b * ld0 + off0 + u] = hb;
  if (hdst1) hdst1[b * ld1 + off1 + u] = hb;
}

// energy[b][t] = sum_q query[b][q] * key[b][q][t]   (memory-bound, f32)
__global__ void energy_kernel(const float* __restrict__ query,  // [64][128]
                              const float* __restrict__ key,    // [64][128][2048]
                              float* __restrict__ energy) {     // [64][2048]
  int t = blockIdx.x * blockDim.x + threadIdx.x;
  int b = blockIdx.y;
  const float* kb = key + ((size_t)b * QB) * TB + t;
  const float* qb = query + b * QB;
  float acc = 0.f;
#pragma unroll 4
  for (int q = 0; q < QB; ++q) acc += qb[q] * kb[(size_t)q * TB];
  energy[b * TB + t] = acc;
}

// Masked softmax with p=1 renormalization (== masked softmax). One block per b.
__global__ void softmax_kernel(const float* __restrict__ energy,
                               const int* __restrict__ seq_sizes,
                               float* __restrict__ score_ws,    // [64][2048]
                               float* __restrict__ out_scores,  // d_out [B][S][T]
                               int s) {
  __shared__ float red[256];
  int b = blockIdx.x, tid = threadIdx.x;
  int n = seq_sizes[b];
  const float* e = energy + b * TB;
  float mx = -3.4e38f;
  for (int t = tid; t < TB; t += 256) if (t < n) mx = fmaxf(mx, e[t]);
  red[tid] = mx; __syncthreads();
  for (int w = 128; w > 0; w >>= 1) {
    if (tid < w) red[tid] = fmaxf(red[tid], red[tid + w]);
    __syncthreads();
  }
  mx = red[0]; __syncthreads();
  float sum = 0.f;
  for (int t = tid; t < TB; t += 256) if (t < n) sum += __expf(e[t] - mx);
  red[tid] = sum; __syncthreads();
  for (int w = 128; w > 0; w >>= 1) {
    if (tid < w) red[tid] += red[tid + w];
    __syncthreads();
  }
  float inv = 1.f / fmaxf(red[0], 1e-12f);
  float* od = out_scores + ((size_t)b * SB + s) * TB;
  float* sw = score_ws + b * TB;
  for (int t = tid; t < TB; t += 256) {
    float v = (t < n) ? __expf(e[t] - mx) * inv : 0.f;
    od[t] = v;
    sw[t] = v;
  }
}

// Partial context: one block per (T-chunk of 256, b); 128 threads = CTX lanes.
__global__ void context_partial_kernel(const float* __restrict__ score,  // [64][2048]
                                       const float* __restrict__ value,  // [64][2048][128]
                                       float* __restrict__ partial) {    // [64][8][128]
  int c = threadIdx.x, chunk = blockIdx.x, b = blockIdx.y;
  const float* sb = score + b * TB + chunk * 256;
  const float* vb = value + ((size_t)b * TB + chunk * 256) * CTXB + c;
  float acc = 0.f;
  for (int t = 0; t < 256; ++t) acc += sb[t] * vb[(size_t)t * CTXB];
  partial[(b * 8 + chunk) * CTXB + c] = acc;
}

// Reduce partials; write bf16 ctx into featbuf (this step's MLP) and
// xcat0 (next step's LSTM-0 input).
__global__ void context_reduce_kernel(const float* __restrict__ partial,
                                      unsigned short* __restrict__ featbuf,
                                      unsigned short* __restrict__ xcat0) {
  int idx = blockIdx.x * blockDim.x + threadIdx.x;  // 64*128
  int b = idx >> 7, c = idx & 127;
  float acc = 0.f;
#pragma unroll
  for (int k = 0; k < 8; ++k) acc += partial[(b * 8 + k) * CTXB + c];
  unsigned short v = f2bf(acc);
  featbuf[b * 384 + 256 + c] = v;
  xcat0[b * 640 + 256 + c] = v;
}

// Classifier (V=33), greedy argmax, and embedding feedback for next step.
__global__ void predict_kernel(const float* __restrict__ hidden,  // [64][256]
                               const float* __restrict__ Ws2,     // [33][256]
                               const float* __restrict__ bs2,
                               const float* __restrict__ embed,   // [33][256]
                               float* __restrict__ out_pred,      // d_out + B*S*T
                               unsigned short* __restrict__ xcat0,
                               int s) {
  __shared__ float pv[VB];
  __shared__ int best;
  int b = blockIdx.x, tid = threadIdx.x;  // 128 threads
  if (tid < VB) {
    const float* h = hidden + b * HB;
    const float* w = Ws2 + tid * HB;
    float acc = bs2[tid];
    for (int k = 0; k < HB; ++k) acc += h[k] * w[k];
    pv[tid] = acc;
    out_pred[((size_t)b * SB + s) * VB + tid] = acc;
  }
  __syncthreads();
  if (tid == 0) {
    int bi = 0; float bv = pv[0];
    for (int v = 1; v < VB; ++v) if (pv[v] > bv) { bv = pv[v]; bi = v; }
    best = bi;
  }
  __syncthreads();
  const float* er = embed + best * HB;
  for (int k = tid; k < HB; k += 128) xcat0[b * 640 + k] = f2bf(er[k]);
}

// ---- one-time setup kernels ----

__global__ void cat_weights_kernel(const float* __restrict__ Wih,
                                   const float* __restrict__ Whh,
                                   unsigned short* __restrict__ out,
                                   int Kih, int total) {
  int idx = blockIdx.x * blockDim.x + threadIdx.x;
  if (idx >= total) return;
  int Kt = Kih + HB;
  int n = idx / Kt, k = idx % Kt;
  float v = (k < Kih) ? Wih[n * Kih + k] : Whh[n * HB + (k - Kih)];
  out[idx] = f2bf(v);
}

__global__ void f32_to_bf16_kernel(const float* __restrict__ in,
                                   unsigned short* __restrict__ out, int n) {
  int idx = blockIdx.x * blockDim.x + threadIdx.x;
  if (idx < n) out[idx] = f2bf(in[idx]);
}

__global__ void add_bias_kernel(const float* __restrict__ a,
                                const float* __restrict__ b,
                                float* __restrict__ out, int n) {
  int idx = blockIdx.x * blockDim.x + threadIdx.x;
  if (idx < n) out[idx] = a[idx] + b[idx];
}

// Initialize xcat0=[embed[0]|con|h0_0], layer state slots, and c states.
__global__ void init_states_kernel(const float* __restrict__ embed,
                                   const float* __restrict__ con,
                                   const float* __restrict__ h0,
                                   const float* __restrict__ c0,
                                   unsigned short* __restrict__ xcat0,
                                   unsigned short* __restrict__ xcat1,
                                   unsigned short* __restrict__ xcat2,
                                   float* __restrict__ cstates) {
  int idx = blockIdx.x * blockDim.x + threadIdx.x;  // 64*640
  if (idx >= BB * 640) return;
  int b = idx / 640, k = idx % 640;
  float v = (k < 256) ? embed[k] : (k < 384 ? con[k - 256] : h0[k - 384]);
  xcat0[idx] = f2bf(v);
  if (k < 256) {
    xcat1[b * 512 + 256 + k] = f2bf(h0[256 + k]);
    xcat2[b * 512 + 256 + k] = f2bf(h0[512 + k]);
    xcat1[b * 512 + k] = 0;
    xcat2[b * 512 + k] = 0;
    cstates[(0 * BB + b) * HB + k] = c0[k];
    cstates[(1 * BB + b) * HB + k] = c0[256 + k];
    cstates[(2 * BB + b) * HB + k] = c0[512 + k];
  }
}

// ---------------------------------------------------------------------------

static inline size_t align256(size_t x) { return (x + 255) & ~(size_t)255; }

extern "C" void kernel_launch(void* const* d_in, const int* in_sizes, int n_in,
                              void* d_out, int out_size, void* d_ws, size_t ws_size,
                              hipStream_t stream) {
  (void)in_sizes; (void)n_in; (void)out_size; (void)ws_size;

  const int*   seq_sizes = (const int*)d_in[0];
  const float* att_key   = (const float*)d_in[1];   // [64][128][2048]
  const float* att_value = (const float*)d_in[2];   // [64][2048][128]
  const float* embed     = (const float*)d_in[4];   // [33][256]
  const float* Wih0 = (const float*)d_in[5];
  const float* Whh0 = (const float*)d_in[6];
  const float* bih0 = (const float*)d_in[7];
  const float* bhh0 = (const float*)d_in[8];
  const float* Wih1 = (const float*)d_in[9];
  const float* Whh1 = (const float*)d_in[10];
  const float* bih1 = (const float*)d_in[11];
  const float* bhh1 = (const float*)d_in[12];
  const float* Wih2 = (const float*)d_in[13];
  const float* Whh2 = (const float*)d_in[14];
  const float* bih2 = (const float*)d_in[15];
  const float* bhh2 = (const float*)d_in[16];
  const float* Wq  = (const float*)d_in[17];
  const float* bq  = (const float*)d_in[18];
  const float* Ws1 = (const float*)d_in[19];
  const float* bs1 = (const float*)d_in[20];
  const float* Ws2 = (const float*)d_in[21];
  const float* bs2 = (const float*)d_in[22];
  const float* h0  = (const float*)d_in[23];
  const float* c0  = (const float*)d_in[24];
  const float* con = (const float*)d_in[25];

  float* out_scores = (float*)d_out;                                  // [64][220][2048]
  float* out_pred   = (float*)d_out + (size_t)BB * SB * TB;           // [64][220][33]

  // ---- workspace layout (bf16 stored as unsigned short) ----
  char* ws = (char*)d_ws;
  size_t off = 0;
  unsigned short* Wcat0 = (unsigned short*)(ws + off); off = align256(off + 1024u * 640 * 2);
  unsigned short* Wcat1 = (unsigned short*)(ws + off); off = align256(off + 1024u * 512 * 2);
  unsigned short* Wcat2 = (unsigned short*)(ws + off); off = align256(off + 1024u * 512 * 2);
  unsigned short* Wqbf  = (unsigned short*)(ws + off); off = align256(off + (size_t)QB * HB * 2);
  unsigned short* Ws1bf = (unsigned short*)(ws + off); off = align256(off + (size_t)HB * 384 * 2);
  float* bcat0 = (float*)(ws + off); off = align256(off + 1024 * 4);
  float* bcat1 = (float*)(ws + off); off = align256(off + 1024 * 4);
  float* bcat2 = (float*)(ws + off); off = align256(off + 1024 * 4);
  unsigned short* xcat0   = (unsigned short*)(ws + off); off = align256(off + (size_t)BB * 640 * 2);
  unsigned short* xcat1   = (unsigned short*)(ws + off); off = align256(off + (size_t)BB * 512 * 2);
  unsigned short* xcat2   = (unsigned short*)(ws + off); off = align256(off + (size_t)BB * 512 * 2);
  unsigned short* featbuf = (unsigned short*)(ws + off); off = align256(off + (size_t)BB * 384 * 2);
  float* cstates = (float*)(ws + off); off = align256(off + (size_t)3 * BB * HB * 4);
  float* gates   = (float*)(ws + off); off = align256(off + (size_t)BB * 1024 * 4);
  float* queryf  = (float*)(ws + off); off = align256(off + (size_t)BB * QB * 4);
  float* hidden  = (float*)(ws + off); off = align256(off + (size_t)BB * HB * 4);
  float* energy  = (float*)(ws + off); off = align256(off + (size_t)BB * TB * 4);
  float* score   = (float*)(ws + off); off = align256(off + (size_t)BB * TB * 4);
  float* partial = (float*)(ws + off); off = align256(off + (size_t)BB * 8 * CTXB * 4);

  // ---- setup: weight conversion / concatenation, biases, state init ----
  {
    int t0 = 1024 * 640, t1 = 1024 * 512;
    cat_weights_kernel<<<(t0 + 255) / 256, 256, 0, stream>>>(Wih0, Whh0, Wcat0, 384, t0);
    cat_weights_kernel<<<(t1 + 255) / 256, 256, 0, stream>>>(Wih1, Whh1, Wcat1, 256, t1);
    cat_weights_kernel<<<(t1 + 255) / 256, 256, 0, stream>>>(Wih2, Whh2, Wcat2, 256, t1);
    f32_to_bf16_kernel<<<(QB * HB + 255) / 256, 256, 0, stream>>>(Wq, Wqbf, QB * HB);
    f32_to_bf16_kernel<<<(HB * 384 + 255) / 256, 256, 0, stream>>>(Ws1, Ws1bf, HB * 384);
    add_bias_kernel<<<4, 256, 0, stream>>>(bih0, bhh0, bcat0, 1024);
    add_bias_kernel<<<4, 256, 0, stream>>>(bih1, bhh1, bcat1, 1024);
    add_bias_kernel<<<4, 256, 0, stream>>>(bih2, bhh2, bcat2, 1024);
    init_states_kernel<<<(BB * 640 + 255) / 256, 256, 0, stream>>>(
        embed, con, h0, c0, xcat0, xcat1, xcat2, cstates);
  }

  // ---- sequential decode: S = 220 steps ----
  for (int s = 0; s < SB; ++s) {
    // LSTM layer 0: gates = [emb|ctx|h0] @ [Wih0|Whh0]^T + b   (K=640)
    gemm_bf16_wmma<<<64, 128, 0, stream>>>(xcat0, 640, Wcat0, 640,
                                           gates, 1024, bcat0, 64, 640, 0);
    lstm_cell_kernel<<<64, 256, 0, stream>>>(gates, cstates + 0 * BB * HB,
                                             xcat0, 640, 384, xcat1, 512, 0);
    // LSTM layer 1 (K=512)
    gemm_bf16_wmma<<<64, 128, 0, stream>>>(xcat1, 512, Wcat1, 512,
                                           gates, 1024, bcat1, 64, 512, 0);
    lstm_cell_kernel<<<64, 256, 0, stream>>>(gates, cstates + 1 * BB * HB,
                                             xcat1, 512, 256, xcat2, 512, 0);
    // LSTM layer 2 (K=512); h2 goes to featbuf[:,0:256]
    gemm_bf16_wmma<<<64, 128, 0, stream>>>(xcat2, 512, Wcat2, 512,
                                           gates, 1024, bcat2, 64, 512, 0);
    lstm_cell_kernel<<<64, 256, 0, stream>>>(gates, cstates + 2 * BB * HB,
                                             xcat2, 512, 256, featbuf, 384, 0);
    // query = h2 @ Wq^T + bq   (A = featbuf[:,0:256], lda=384, K=256)
    gemm_bf16_wmma<<<8, 128, 0, stream>>>(featbuf, 384, Wqbf, 256,
                                          queryf, 128, bq, 8, 256, 0);
    // attention energy + masked softmax + context
    energy_kernel<<<dim3(TB / 256, BB), 256, 0, stream>>>(queryf, att_key, energy);
    softmax_kernel<<<BB, 256, 0, stream>>>(energy, seq_sizes, score, out_scores, s);
    context_partial_kernel<<<dim3(8, BB), 128, 0, stream>>>(score, att_value, partial);
    context_reduce_kernel<<<(BB * CTXB) / 256, 256, 0, stream>>>(partial, featbuf, xcat0);
    // hidden = relu([h2|ctx] @ Ws1^T + bs1)   (K=384)
    gemm_bf16_wmma<<<16, 128, 0, stream>>>(featbuf, 384, Ws1bf, 384,
                                           hidden, 256, bs1, 16, 384, 1);
    // predict + argmax + embedding feedback
    predict_kernel<<<BB, 128, 0, stream>>>(hidden, Ws2, bs2, embed,
                                           out_pred, xcat0, s);
  }
}
